// PDN_80032420594375
// MI455X (gfx1250) — matrix operand order repaired and checked by reading, not compiled, not run on hardware
//
#include <hip/hip_runtime.h>
#include <hip/hip_bf16.h>
#include <stdint.h>

// ---------------- problem constants (from reference) ----------------
#define C_CH   3072
#define C_L4   2048
#define C_L3   1024
#define Hh     28
#define Ww     28
#define HW     784      // 28*28
#define Bsz    16
#define L1     9        // landmarks + 1
#define Kcls   2000
#define Ncols  (Bsz * L1)   // 144 GEMM columns
#define NW     (Ncols / 2)  // 72 packed-pair words per k-row

typedef __attribute__((ext_vector_type(16))) __bf16 v16bf;
typedef __attribute__((ext_vector_type(8)))  float  v8f;

// d_out layout: [0,32000) class_scores | [32000,144896) maps | [144896,587264) all_features
#define OUT_MAPS_OFF 32000
#define OUT_AF_OFF   144896

// ws layout (bytes)
#define WS_ASQ     0
#define WS_WB      256                         // W_class bf16: 2000*3072*2 = 12,288,000
#define WS_FB      (256 + 12288000)            // packed features bf16: 3072*144*2 = 884,736
#define WS_SCORES  13173248                    // scores fp32: 2000*144*4 = 1,152,000

// ---------------- x(b,c,h,w) on the fly: concat(bilinear-up(l4), l3) ----------------
__device__ __forceinline__ float load_x(const float* __restrict__ l3,
                                        const float* __restrict__ l4,
                                        int b, int c, int hh, int ww) {
  if (c >= C_L4) {
    return l3[(((size_t)b * C_L3 + (c - C_L4)) * Hh + hh) * Ww + ww];
  }
  // bilinear 14x14 -> 28x28, half-pixel centers, edge clamp
  float sy = hh * 0.5f - 0.25f;
  float sx = ww * 0.5f - 0.25f;
  int y0 = (int)floorf(sy); float fy = sy - (float)y0;
  int x0 = (int)floorf(sx); float fx = sx - (float)x0;
  int y1 = y0 + 1, x1 = x0 + 1;
  y0 = min(max(y0, 0), 13); y1 = min(max(y1, 0), 13);
  x0 = min(max(x0, 0), 13); x1 = min(max(x1, 0), 13);
  const float* base = l4 + ((size_t)b * C_L4 + c) * 196;
  float v00 = base[y0 * 14 + x0], v01 = base[y0 * 14 + x1];
  float v10 = base[y1 * 14 + x0], v11 = base[y1 * 14 + x1];
  return (1.f - fy) * ((1.f - fx) * v00 + fx * v01) +
         fy         * ((1.f - fx) * v10 + fx * v11);
}

// ---------------- kernel 0a: a_sq[l] = ||W_land[l]||^2 ----------------
__global__ void k_asq(const float* __restrict__ Wl, float* __restrict__ asq) {
  int wv = threadIdx.x >> 5, lane = threadIdx.x & 31;
  if (wv >= L1) return;
  float s = 0.f;
  for (int c = lane; c < C_CH; c += 32) {
    float w = Wl[wv * C_CH + c];
    s += w * w;
  }
  #pragma unroll
  for (int off = 16; off >= 1; off >>= 1) s += __shfl_xor(s, off, 32);
  if (lane == 0) asq[wv] = s;
}

// ---------------- kernel 0b: W_class fp32 -> bf16 ----------------
__global__ void k_cvt(const float* __restrict__ src, __bf16* __restrict__ dst, int n) {
  int i = blockIdx.x * blockDim.x + threadIdx.x;
  if (i < n) dst[i] = (__bf16)src[i];
}

// ---------------- kernel 1: landmark distance maps + channel softmax ----------------
// one wave32 per pixel; lane-strided channel accumulation + butterfly reduce
__global__ void k_maps(const float* __restrict__ l3, const float* __restrict__ l4,
                       const float* __restrict__ Wl, const float* __restrict__ asq,
                       float* __restrict__ maps_out) {
  int wave = threadIdx.x >> 5;
  int lane = threadIdx.x & 31;
  int pix  = blockIdx.x * 8 + wave;
  if (pix >= Bsz * HW) return;
  int b  = pix / HW;
  int p  = pix % HW;
  int hh = p / Ww, ww = p % Ww;

  float dot[L1];
  #pragma unroll
  for (int l = 0; l < L1; ++l) dot[l] = 0.f;
  float sq = 0.f;

  for (int c = lane; c < C_CH; c += 32) {
    float xv = load_x(l3, l4, b, c, hh, ww);
    sq += xv * xv;
    #pragma unroll
    for (int l = 0; l < L1; ++l) dot[l] += xv * Wl[l * C_CH + c];
  }
  #pragma unroll
  for (int off = 16; off >= 1; off >>= 1) {
    sq += __shfl_xor(sq, off, 32);
    #pragma unroll
    for (int l = 0; l < L1; ++l) dot[l] += __shfl_xor(dot[l], off, 32);
  }
  // logits = -(||x||^2 - 2 x.w_l + ||w_l||^2)
  float logit[L1], mx = -3.0e38f;
  #pragma unroll
  for (int l = 0; l < L1; ++l) {
    logit[l] = 2.f * dot[l] - sq - asq[l];
    mx = fmaxf(mx, logit[l]);
  }
  float e[L1], sum = 0.f;
  #pragma unroll
  for (int l = 0; l < L1; ++l) { e[l] = __expf(logit[l] - mx); sum += e[l]; }
  float inv = 1.f / sum;
  if (lane < L1) {
    float v = 0.f;
    #pragma unroll
    for (int l = 0; l < L1; ++l) v = (l == lane) ? e[l] * inv : v;
    maps_out[((size_t)b * L1 + lane) * HW + p] = v;
  }
}

// ---------------- kernel 2: weighted spatial pooling -> all_features + bf16 packed copy ----
// one block (128 thr = 4 waves) per (b, c)
__global__ void k_feat(const float* __restrict__ l3, const float* __restrict__ l4,
                       const float* __restrict__ maps, const float* __restrict__ modu,
                       float* __restrict__ af_out, __bf16* __restrict__ Fb) {
  int idx = blockIdx.x;            // b*C + c
  int b = idx / C_CH;
  int c = idx % C_CH;
  int tid = threadIdx.x, lane = tid & 31, wv = tid >> 5;

  float acc[L1];
  #pragma unroll
  for (int l = 0; l < L1; ++l) acc[l] = 0.f;

  const float* mrow = maps + (size_t)b * L1 * HW;
  for (int p = tid; p < HW; p += 128) {
    int hh = p / Ww, ww = p % Ww;
    float xv = load_x(l3, l4, b, c, hh, ww);
    #pragma unroll
    for (int l = 0; l < L1; ++l) acc[l] += xv * mrow[l * HW + p];
  }
  #pragma unroll
  for (int off = 16; off >= 1; off >>= 1) {
    #pragma unroll
    for (int l = 0; l < L1; ++l) acc[l] += __shfl_xor(acc[l], off, 32);
  }
  __shared__ float red[L1][4];
  if (lane == 0) {
    #pragma unroll
    for (int l = 0; l < L1; ++l) red[l][wv] = acc[l];
  }
  __syncthreads();
  if (tid < L1) {
    float s  = red[tid][0] + red[tid][1] + red[tid][2] + red[tid][3];
    float af = s * (1.0f / (float)HW);
    af_out[((size_t)b * C_CH + c) * L1 + tid] = af;           // pre-modulation output
    float mv = af * modu[c * L1 + tid];                       // modulated -> GEMM input
    int n = b * L1 + tid;
    // k-pair packed B layout: bf16 idx = (c&~1)*Ncols + 2n + (c&1)
    Fb[(size_t)(c & ~1) * Ncols + 2 * n + (c & 1)] = (__bf16)mv;
  }
}

// ---------------- kernel 3: scores = W_class_bf16 (2000x3072) @ F_bf16 (3072x144) --------
// one wave per 16x16 tile; 96 steps of v_wmma_f32_16x16x32_bf16 along K=3072
__global__ void k_gemm(const __bf16* __restrict__ Wb, const uint32_t* __restrict__ FbP,
                       float* __restrict__ scores) {
  const int MT = Kcls / 16;    // 125
  const int NT = Ncols / 16;   // 9
  int wave = (blockIdx.x * blockDim.x + threadIdx.x) >> 5;
  int lane = threadIdx.x & 31;
  if (wave >= MT * NT) return;
  int tm = wave / NT, tn = wave % NT;
  int half = lane >> 4;        // 0: lanes 0-15, 1: lanes 16-31
  int m16  = lane & 15;
  int rowA = tm * 16 + m16;    // A: M = lane%16
  int n    = tn * 16 + m16;    // B/C/D: N = lane%16
  const uint4* Arow = (const uint4*)(Wb + (size_t)rowA * C_CH);   // 16B-chunk view of one A row

  v8f acc = (v8f)0.f;
  for (int k0 = 0; k0 < C_CH; k0 += 32) {
    // A 16x32 bf16 layout: lanes<16 take K {k0..k0+7, k0+16..k0+23}; lanes>=16 take {+8..+15, +24..+31}
    union { uint4 q[2]; v16bf v; } A;
    int qlo = (k0 >> 3) + half;
    A.q[0] = Arow[qlo];
    A.q[1] = Arow[qlo + 2];
    __builtin_prefetch((const void*)(Arow + qlo + 8), 0, 1);      // global_prefetch on A stream
    // B 32x16 bf16 layout: lanes<16 K {k0..k0+7, k0+16..k0+23}; lanes>=16 K {+8..+15, +24..+31}; VGPR v = pair (k,k+1)
    union { uint32_t w[8]; v16bf v; } Bm;
    int kb = k0 + 8 * half;
    #pragma unroll
    for (int v2 = 0; v2 < 8; ++v2)
      Bm.w[v2] = FbP[(size_t)(kb + 2 * v2 + (v2 >= 4 ? 8 : 0)) * NW + n];
    acc = __builtin_amdgcn_wmma_f32_16x16x32_bf16(
        /*neg_a=*/false, A.v, /*neg_b=*/false, Bm.v,
        /*c_mod=*/(short)0, acc, /*reuse_a=*/false, /*reuse_b=*/false);
  }
  // C/D layout: VGPR r -> M = r (+8 for lanes>=16), N = lane%16
  #pragma unroll
  for (int r = 0; r < 8; ++r) {
    int m = tm * 16 + r + 8 * half;
    scores[(size_t)m * Ncols + n] = acc[r];
  }
}

// ---------------- kernel 4: class_scores = mean over l=0..7 of scores ----------------
__global__ void k_cls(const float* __restrict__ scores, float* __restrict__ out) {
  int idx = blockIdx.x * blockDim.x + threadIdx.x;
  if (idx >= Bsz * Kcls) return;
  int b = idx / Kcls, k = idx % Kcls;
  const float* s = scores + (size_t)k * Ncols + b * L1;
  float acc = 0.f;
  #pragma unroll
  for (int l = 0; l < 8; ++l) acc += s[l];
  out[idx] = acc * 0.125f;
}

// ---------------- launcher ----------------
extern "C" void kernel_launch(void* const* d_in, const int* in_sizes, int n_in,
                              void* d_out, int out_size, void* d_ws, size_t ws_size,
                              hipStream_t stream) {
  const float* l3 = (const float*)d_in[0];
  const float* l4 = (const float*)d_in[1];
  const float* Wl = (const float*)d_in[2];
  const float* Wc = (const float*)d_in[3];
  const float* mo = (const float*)d_in[4];
  float* out = (float*)d_out;

  char* ws = (char*)d_ws;
  float*  asq    = (float*)(ws + WS_ASQ);
  __bf16* Wb     = (__bf16*)(ws + WS_WB);
  __bf16* Fb     = (__bf16*)(ws + WS_FB);
  float*  scores = (float*)(ws + WS_SCORES);

  // 0a: landmark norms
  k_asq<<<1, 288, 0, stream>>>(Wl, asq);
  // 0b: classifier weights -> bf16
  const int NWC = Kcls * C_CH;
  k_cvt<<<(NWC + 255) / 256, 256, 0, stream>>>(Wc, Wb, NWC);
  // 1: maps (softmax over landmarks), written straight into its output slice
  k_maps<<<(Bsz * HW + 7) / 8, 256, 0, stream>>>(l3, l4, Wl, asq, out + OUT_MAPS_OFF);
  // 2: pooled features (fp32 output) + modulated bf16 packed GEMM operand
  k_feat<<<Bsz * C_CH, 128, 0, stream>>>(l3, l4, out + OUT_MAPS_OFF, mo,
                                         out + OUT_AF_OFF, Fb);
  // 3: classifier GEMM via bf16 WMMA (125x9 = 1125 tiles, 8 waves/block)
  const int tiles = (Kcls / 16) * (Ncols / 16);
  k_gemm<<<(tiles * 32 + 255) / 256, 256, 0, stream>>>(Wb, (const uint32_t*)Fb, scores);
  // 4: mean over first 8 landmarks
  k_cls<<<(Bsz * Kcls + 255) / 256, 256, 0, stream>>>(scores, out);
}
